// Attention_16544214024725
// MI455X (gfx1250) — matrix operand order repaired and checked by reading, not compiled, not run on hardware
//
#include <hip/hip_runtime.h>
#include <hip/hip_bf16.h>

// MI455X / gfx1250, wave32. All matmuls via v_wmma_f32_16x16x32_bf16.
// Workspace requirement: (4096*6144 + 4096*2048) bf16 = 64 MB.

typedef __attribute__((ext_vector_type(16))) __bf16 v16bf;
typedef __attribute__((ext_vector_type(8)))  float  v8f;

__device__ __forceinline__ unsigned short f32_to_bf16(float f) {
    union { float f; unsigned u; } v; v.f = f;
    unsigned r = v.u + 0x7FFFu + ((v.u >> 16) & 1u);   // round to nearest even
    return (unsigned short)(r >> 16);
}

__device__ __forceinline__ float fast_exp2(float x) {
    return __builtin_amdgcn_exp2f(x);                  // v_exp_f32 (TRANS unit)
}

__device__ __forceinline__ v8f wmma_bf16(v16bf a, v16bf b, v8f c) {
    return __builtin_amdgcn_wmma_f32_16x16x32_bf16(
        /*neg_a=*/false, a, /*neg_b=*/false, b,
        /*c_mod=*/(short)0, c, /*reuse_a=*/false, /*reuse_b=*/false);
}

// A-matrix fragment (16x32, 16-bit). Lane L: m = row16 + L%16, h = L/16.
// elements 0..7  -> k = k0 + h*8 + [0..7]
// elements 8..15 -> k = k0 + 16 + h*8 + [0..7]
__device__ __forceinline__ v16bf load_a_frag(const unsigned short* base, int stride,
                                             int row16, int k0) {
    const int lane = threadIdx.x & 31;
    const unsigned short* p = base + (size_t)(row16 + (lane & 15)) * stride
                                   + k0 + ((lane >> 4) << 3);
    v16bf f;
    ((uint4*)&f)[0] = *(const uint4*)(p);
    ((uint4*)&f)[1] = *(const uint4*)(p + 16);
    return f;
}

// B-matrix fragment (32x16, 16-bit). Lane L: n = row16 + L%16, h = L/16.
// elements 0..7  -> k = k0 + h*8 + [0..7]
// elements 8..15 -> k = k0 + 16 + h*8 + [0..7]
__device__ __forceinline__ v16bf load_b_frag(const unsigned short* base, int stride,
                                             int row16, int k0) {
    const int lane = threadIdx.x & 31;
    const unsigned short* p = base + (size_t)(row16 + (lane & 15)) * stride
                                   + k0 + ((lane >> 4) << 3);
    v16bf f;
    ((uint4*)&f)[0] = *(const uint4*)(p);
    ((uint4*)&f)[1] = *(const uint4*)(p + 16);
    return f;
}

// ---------------------------------------------------------------------------
// GEMM: D[M][N] = A[M][K] * B[N][K]^T.  B always fp32 (a weight matrix).
// Block tile 128x128x32, 256 threads = 8 waves (2x4), each wave 64x32.
// Double-buffered LDS: stage tile kk+32 while WMMAs consume tile kk.
// ---------------------------------------------------------------------------
template<bool A_BF16, bool OUT_BF16>
__global__ __launch_bounds__(256)
void gemm_bt_wmma(const void* __restrict__ Ap, const float* __restrict__ B,
                  void* __restrict__ Dp, int M, int N, int K) {
    constexpr int BM = 128, BN = 128, BK = 32;
    constexpr int LDK = 48;                       // row stride (ushort), 96 B
    __shared__ alignas(16) unsigned short sA[2][BM * LDK];
    __shared__ alignas(16) unsigned short sB[2][BN * LDK];

    const int tid  = threadIdx.x;
    const int lane = tid & 31;
    const int wid  = tid >> 5;        // 0..7
    const int wrow = wid >> 2;        // 0..1  -> 64 rows
    const int wcol = wid & 3;         // 0..3  -> 32 cols
    const int block_m = blockIdx.y * BM;
    const int block_n = blockIdx.x * BN;

    const float*          Af  = (const float*)Ap;
    const unsigned short* Abf = (const unsigned short*)Ap;

    auto stage = [&](int kk, int buf) {
        // ---- A tile ----
        if constexpr (A_BF16) {
            #pragma unroll
            for (int i = tid; i < BM * BK / 8; i += 256) {
                int m = i >> 2;
                int c = (i & 3) << 3;
                const unsigned short* gp = Abf + (size_t)(block_m + m) * K + kk + c;
                *(uint4*)&sA[buf][m * LDK + c] = *(const uint4*)gp;
                if (kk + BK < K) __builtin_prefetch(gp + BK, 0, 0);
            }
        } else {
            #pragma unroll
            for (int i = tid; i < BM * BK / 4; i += 256) {
                int m = i >> 3;
                int c = (i & 7) << 2;
                const float* gp = Af + (size_t)(block_m + m) * K + kk + c;
                float4 v = *(const float4*)gp;
                ushort4 u;
                u.x = f32_to_bf16(v.x); u.y = f32_to_bf16(v.y);
                u.z = f32_to_bf16(v.z); u.w = f32_to_bf16(v.w);
                *(ushort4*)&sA[buf][m * LDK + c] = u;
                if (kk + BK < K) __builtin_prefetch(gp + BK, 0, 0);
            }
        }
        // ---- B tile (fp32 -> bf16) ----
        #pragma unroll
        for (int i = tid; i < BN * BK / 4; i += 256) {
            int n = i >> 3;
            int c = (i & 7) << 2;
            const float* gp = B + (size_t)(block_n + n) * K + kk + c;
            float4 v = *(const float4*)gp;
            ushort4 u;
            u.x = f32_to_bf16(v.x); u.y = f32_to_bf16(v.y);
            u.z = f32_to_bf16(v.z); u.w = f32_to_bf16(v.w);
            *(ushort4*)&sB[buf][n * LDK + c] = u;
            if (kk + BK < K) __builtin_prefetch(gp + BK, 0, 0);
        }
    };

    const v8f vzero = {0.f,0.f,0.f,0.f,0.f,0.f,0.f,0.f};
    v8f acc[4][2];
    #pragma unroll
    for (int i = 0; i < 4; ++i) for (int j = 0; j < 2; ++j) acc[i][j] = vzero;

    stage(0, 0);
    __syncthreads();
    int cur = 0;
    for (int kk = 0; kk < K; kk += BK) {
        if (kk + BK < K) stage(kk + BK, cur ^ 1);   // overlaps with WMMAs below
        v16bf af[4];
        #pragma unroll
        for (int i = 0; i < 4; ++i) af[i] = load_a_frag(sA[cur], LDK, wrow * 64 + i * 16, 0);
        v16bf bf2[2];
        #pragma unroll
        for (int j = 0; j < 2; ++j) bf2[j] = load_b_frag(sB[cur], LDK, wcol * 32 + j * 16, 0);
        #pragma unroll
        for (int i = 0; i < 4; ++i)
            #pragma unroll
            for (int j = 0; j < 2; ++j)
                acc[i][j] = wmma_bf16(af[i], bf2[j], acc[i][j]);
        __syncthreads();
        cur ^= 1;
    }

    // ---- store D (C/D layout: m = (lane/16)*8 + r, n = lane%16) ----
    float*          Df  = (float*)Dp;
    unsigned short* Dbf = (unsigned short*)Dp;
    #pragma unroll
    for (int i = 0; i < 4; ++i)
        #pragma unroll
        for (int j = 0; j < 2; ++j) {
            int m0 = block_m + wrow * 64 + i * 16 + ((lane >> 4) << 3);
            int n  = block_n + wcol * 32 + j * 16 + (lane & 15);
            #pragma unroll
            for (int r = 0; r < 8; ++r) {
                float v = acc[i][j][r];
                if constexpr (OUT_BF16) Dbf[(size_t)(m0 + r) * N + n] = f32_to_bf16(v);
                else                    Df [(size_t)(m0 + r) * N + n] = v;
            }
        }
}

// ---------------------------------------------------------------------------
// Flash attention. qkv (bf16): [B*S][3*H*D] rows; combined (bf16): [B*S][H*D].
// One block = (b, h, 128-query tile); 8 waves, each wave owns 16 query rows.
// ---------------------------------------------------------------------------
__global__ __launch_bounds__(256)
void attn_kernel(const unsigned short* __restrict__ qkv,
                 unsigned short* __restrict__ combined) {
    constexpr int S = 2048, D = 128, H = 16;
    constexpr int TQ = 128, TK = 64;
    constexpr int QKV_STRIDE = 3 * H * D;           // 6144
    constexpr int LV = TK + 8;                      // 72 ushorts = 144 B
    constexpr int LP = TK + 8;

    __shared__ alignas(16) unsigned short sV[D * LV];       // V^T: [d][key]
    __shared__ alignas(16) unsigned short sP[8 * 16 * LP];  // per-wave P tiles

    const int tid  = threadIdx.x;
    const int lane = tid & 31;
    const int wid  = tid >> 5;
    const int qtile = blockIdx.x;          // 0..15
    const int bh    = blockIdx.y;          // 0..31
    const int b     = bh >> 4;
    const int h     = bh & 15;
    const size_t row0 = (size_t)b * S;

    const unsigned short* qbase = qkv + row0 * QKV_STRIDE + (size_t)h * D;
    const unsigned short* kbase = qkv + row0 * QKV_STRIDE + (size_t)(H + h) * D;
    const unsigned short* vbase = qkv + row0 * QKV_STRIDE + (size_t)(2 * H + h) * D;

    const int q0 = qtile * TQ + wid * 16;  // wave's query row base within head

    // Q fragments straight from global (A-frag layout is 16B-contiguous in qkv rows)
    v16bf qf[4];
    #pragma unroll
    for (int s = 0; s < 4; ++s)
        qf[s] = load_a_frag(qbase, QKV_STRIDE, q0, s * 32);

    const v8f vzero = {0.f,0.f,0.f,0.f,0.f,0.f,0.f,0.f};
    v8f o[8];
    float mrow[8], lrow[8];
    #pragma unroll
    for (int t = 0; t < 8; ++t) o[t] = vzero;
    #pragma unroll
    for (int r = 0; r < 8; ++r) { mrow[r] = -3.0e38f; lrow[r] = 0.f; }

    // base-2 softmax domain: logits2 = S * (scale * log2e); p = exp2(logits2 - max2)
    const float scale2 = 0.08838834764831845f * 1.4426950408889634f;

    for (int kt = 0; kt < S; kt += TK) {
        __syncthreads();   // protect sV from previous iteration's readers
        // stage V^T into LDS: [d][key]
        #pragma unroll
        for (int i = tid; i < TK * D / 8; i += 256) {
            int key = i >> 4;                 // D/8 = 16 chunks per key row
            int d8  = (i & 15) << 3;
            uint4 chunk = *(const uint4*)(vbase + (size_t)(kt + key) * QKV_STRIDE + d8);
            unsigned short tmp[8]; *(uint4*)tmp = chunk;
            #pragma unroll
            for (int e = 0; e < 8; ++e) sV[(d8 + e) * LV + key] = tmp[e];
        }
        __syncthreads();

        // ---- S = Q K^T, 16x64 per wave. Preload ALL 16 K fragments into
        //      distinct registers (one 32-load clause, s-major issue order),
        //      then run 16 WMMAs whose loadcnt waits stagger downward. ----
        v16bf kf[4][4];                       // [s][nt]
        #pragma unroll
        for (int s = 0; s < 4; ++s)
            #pragma unroll
            for (int nt = 0; nt < 4; ++nt)
                kf[s][nt] = load_b_frag(kbase, QKV_STRIDE, kt + nt * 16, s * 32);

        v8f sacc[4];
        #pragma unroll
        for (int nt = 0; nt < 4; ++nt) sacc[nt] = vzero;
        #pragma unroll
        for (int s = 0; s < 4; ++s)           // consume in issue order
            #pragma unroll
            for (int nt = 0; nt < 4; ++nt)
                sacc[nt] = wmma_bf16(qf[s], kf[s][nt], sacc[nt]);

        #pragma unroll
        for (int nt = 0; nt < 4; ++nt)
            #pragma unroll
            for (int r = 0; r < 8; ++r) sacc[nt][r] *= scale2;

        // ---- online softmax (row = (lane/16)*8 + r, 16-lane reductions) ----
        #pragma unroll
        for (int r = 0; r < 8; ++r) {
            float mx = -3.0e38f;
            #pragma unroll
            for (int nt = 0; nt < 4; ++nt) mx = fmaxf(mx, sacc[nt][r]);
            #pragma unroll
            for (int off = 1; off < 16; off <<= 1)
                mx = fmaxf(mx, __shfl_xor(mx, off, 32));
            float mnew = fmaxf(mrow[r], mx);
            float corr = fast_exp2(mrow[r] - mnew);
            lrow[r] *= corr;
            #pragma unroll
            for (int t = 0; t < 8; ++t) o[t][r] *= corr;
            mrow[r] = mnew;

            float rsum = 0.f;
            const int mloc = ((lane >> 4) << 3) + r;
            #pragma unroll
            for (int nt = 0; nt < 4; ++nt) {
                float pv = fast_exp2(sacc[nt][r] - mnew);
                rsum += pv;
                sP[(wid * 16 + mloc) * LP + nt * 16 + (lane & 15)] = f32_to_bf16(pv);
            }
            #pragma unroll
            for (int off = 1; off < 16; off <<= 1)
                rsum += __shfl_xor(rsum, off, 32);
            lrow[r] += rsum;
        }

        // ---- O += P @ V: preload the 8 V^T fragments per k-step as one LDS
        //      clause, then 8 WMMAs ----
        #pragma unroll
        for (int s = 0; s < 2; ++s) {
            v16bf pf = load_a_frag(sP, LP, wid * 16, s * 32);
            v16bf vf[8];
            #pragma unroll
            for (int t = 0; t < 8; ++t) vf[t] = load_b_frag(sV, LV, t * 16, s * 32);
            #pragma unroll
            for (int t = 0; t < 8; ++t) o[t] = wmma_bf16(pf, vf[t], o[t]);
        }
    }

    // ---- epilogue: O /= l, write combined bf16 ----
    #pragma unroll
    for (int r = 0; r < 8; ++r) {
        float inv = 1.0f / lrow[r];
        int m = q0 + ((lane >> 4) << 3) + r;
        size_t out_row = row0 + m;
        #pragma unroll
        for (int t = 0; t < 8; ++t) {
            int d = t * 16 + (lane & 15);
            combined[out_row * (size_t)(H * D) + h * D + d] = f32_to_bf16(o[t][r] * inv);
        }
    }
}

// ---------------------------------------------------------------------------
extern "C" void kernel_launch(void* const* d_in, const int* in_sizes, int n_in,
                              void* d_out, int out_size, void* d_ws, size_t ws_size,
                              hipStream_t stream) {
    const float* x     = (const float*)d_in[0];   // [2,2048,2048]
    const float* w_qkv = (const float*)d_in[1];   // [6144,2048]
    const float* w_out = (const float*)d_in[2];   // [2048,2048]
    float*       out   = (float*)d_out;           // [2,2048,2048]

    const int M = 2 * 2048;        // B*S
    const int NQKV = 6144;         // 3*H*D
    const int DIM = 2048;

    unsigned short* qkv_bf  = (unsigned short*)d_ws;                 // 48 MB
    unsigned short* comb_bf = qkv_bf + (size_t)M * NQKV;             // 16 MB

    dim3 blk(256);
    // 1) qkv = x @ w_qkv^T  (fp32 in, bf16 out to workspace)
    gemm_bt_wmma<false, true><<<dim3(NQKV / 128, M / 128), blk, 0, stream>>>(
        (const void*)x, w_qkv, (void*)qkv_bf, M, NQKV, DIM);
    // 2) flash attention per (b, h, q-tile)
    attn_kernel<<<dim3(2048 / 128, 2 * 16), blk, 0, stream>>>(qkv_bf, comb_bf);
    // 3) out = combined @ w_out^T  (bf16 A, fp32 out)
    gemm_bt_wmma<true, false><<<dim3(DIM / 128, M / 128), blk, 0, stream>>>(
        (const void*)comb_bf, w_out, (void*)out, M, DIM, DIM);
}
